// SelfAttention_68925635166788
// MI455X (gfx1250) — matrix operand order repaired and checked by reading, not compiled, not run on hardware
//
#include <hip/hip_runtime.h>
#include <stdint.h>

// ---------------------------------------------------------------------------
// MI455X (gfx1250) self-attention-with-degenerate-diagonal kernel set.
// All matmuls use v_wmma_f32_16x16x32_bf16 (bf16 in, fp32 accumulate).
// GEMMs use gfx1250 async memory->LDS copies (ASYNCcnt) with double buffering.
// Shapes fixed by the reference: B=4, T=2048, D=2048, N=16 heads, H=128.
// ---------------------------------------------------------------------------

typedef __attribute__((ext_vector_type(16))) __bf16 v16bf;
typedef __attribute__((ext_vector_type(8)))  __bf16 v8bf;
typedef __attribute__((ext_vector_type(8)))  float  v8f;

#define Tt 2048
#define D3 6144

static __device__ __forceinline__ unsigned short f2bf(float f) {
  union { float f; unsigned u; } v; v.f = f;
  unsigned r = v.u + 0x7FFFu + ((v.u >> 16) & 1u);   // round-to-nearest-even
  return (unsigned short)(r >> 16);
}
static __device__ __forceinline__ float bf2f(unsigned short h) {
  union { unsigned u; float f; } v; v.u = ((unsigned)h) << 16; return v.f;
}

// Assemble a v16bf A/B fragment from two 16-byte chunks (per ISA VGPR layout:
// elements 0..7 = first K-chunk, 8..15 = second K-chunk of this lane).
static __device__ __forceinline__ v16bf load_frag(const unsigned short* p1,
                                                  const unsigned short* p2) {
  v8bf a = *(const v8bf*)p1;
  v8bf b = *(const v8bf*)p2;
  return __builtin_shufflevector(a, b, 0,1,2,3,4,5,6,7,8,9,10,11,12,13,14,15);
}

// gfx1250 async global->LDS copy, 16B per lane, tracked by ASYNCcnt.
// lds_off is the wave-relative LDS byte address (low 32 bits of a shared ptr).
static __device__ __forceinline__ void async_load_b128(unsigned lds_off,
                                                       unsigned long long gaddr) {
  asm volatile("global_load_async_to_lds_b128 %0, %1, off"
               :: "v"(lds_off), "v"(gaddr) : "memory");
}
static __device__ __forceinline__ void wait_async0() {
  asm volatile("s_wait_asynccnt 0x0" ::: "memory");
}
static __device__ __forceinline__ unsigned lds_addr(const void* p) {
  return (unsigned)(size_t)p;   // low 32 bits of shared aperture = LDS offset
}

// --------------------------- elementwise converts ---------------------------

__global__ __launch_bounds__(256)
void cvt_f32_bf16(const float* __restrict__ in, unsigned short* __restrict__ out, int n) {
  int i = blockIdx.x * 256 + threadIdx.x;
  if (i < n) out[i] = f2bf(in[i]);
}

// (R x C) fp32 -> (C x R) bf16, LDS-tiled so both sides are coalesced.
__global__ __launch_bounds__(256)
void transpose_f32_to_bf16(const float* __restrict__ in, unsigned short* __restrict__ out,
                           int R, int C) {
  __shared__ float tile[32][33];
  const int c0 = blockIdx.x * 32, r0 = blockIdx.y * 32;
  const int tx = threadIdx.x, ty = threadIdx.y;            // block (32,8)
  #pragma unroll
  for (int j = 0; j < 32; j += 8)
    tile[ty + j][tx] = in[(size_t)(r0 + ty + j) * C + c0 + tx];
  __syncthreads();
  #pragma unroll
  for (int j = 0; j < 32; j += 8)
    out[(size_t)(c0 + ty + j) * R + r0 + tx] = f2bf(tile[tx][ty + j]);
}

// ------------------------------- tiled GEMM --------------------------------
// C(M,N) = A(M,K) * Bt(N,K)^T.  A,Bt bf16 row-major; out fp32 (+bias) or bf16.
// Block: 128 threads = 4 waves; tile 128x128x32; each wave owns 64x64
// (4x4 WMMA tiles, 128 accumulator VGPRs).  Tiles arrive via async->LDS
// (no VGPR staging), double-buffered: the async DMA fills tile k+1 while the
// WMMAs consume tile k; one barrier per k-step.

#define LPITCH 40   // 32 K elems + 8 pad (bank spread)

template<int OUT_BF16, int HAS_BIAS>
__global__ __launch_bounds__(128)
void gemm_bf16_wmma(const unsigned short* __restrict__ A,
                    const unsigned short* __restrict__ Bt,
                    float* __restrict__ Cf,
                    unsigned short* __restrict__ Cb,
                    const float* __restrict__ bias,
                    int K, int Nsz) {
  __shared__ unsigned short lA[2][128 * LPITCH];
  __shared__ unsigned short lB[2][128 * LPITCH];
  const int tid  = threadIdx.x;
  const int lane = tid & 31, half = (tid >> 4) & 1, wv = tid >> 5;
  const int wm = (wv >> 1) * 64, wn = (wv & 1) * 64;
  const int m0 = blockIdx.y * 128, n0 = blockIdx.x * 128;

  v8f acc[16] = {};

  const int row2 = tid >> 1;            // 64 rows per pass, 2 passes
  const int c16  = (tid & 1) * 16;      // which 16-elem half of the 32-wide row

  // Issue one 128x32 A-tile + 128x32 B-tile into LDS buffer `buf` (8 async
  // b128 instructions per wave -> ASYNCcnt tracks them; loads land in-order).
  auto issue_tile = [&](int k0, int buf) {
    #pragma unroll
    for (int rr = 0; rr < 128; rr += 64) {
      const int r = row2 + rr;
      const size_t ga = (size_t)(m0 + r) * K + k0 + c16;
      const size_t gb = (size_t)(n0 + r) * K + k0 + c16;
      const int    lo = r * LPITCH + c16;
      async_load_b128(lds_addr(&lA[buf][lo]),     (unsigned long long)(A  + ga));
      async_load_b128(lds_addr(&lA[buf][lo + 8]), (unsigned long long)(A  + ga + 8));
      async_load_b128(lds_addr(&lB[buf][lo]),     (unsigned long long)(Bt + gb));
      async_load_b128(lds_addr(&lB[buf][lo + 8]), (unsigned long long)(Bt + gb + 8));
    }
  };

  const int KT = K >> 5;
  issue_tile(0, 0);

  for (int kt = 0; kt < KT; ++kt) {
    wait_async0();        // this wave's tile-kt loads have landed in LDS
    __syncthreads();      // everyone's tile-kt landed AND everyone done with kt-1
    if (kt + 1 < KT) issue_tile((kt + 1) << 5, (kt + 1) & 1);  // fill other buffer

    const unsigned short* cA = lA[kt & 1];
    const unsigned short* cB = lB[kt & 1];

    v16bf af[4], bfv[4];
    #pragma unroll
    for (int mt = 0; mt < 4; ++mt) {
      const int rl = (wm + mt * 16 + (lane & 15)) * LPITCH;
      // A 16x32: elems 0..7 -> K = half*8+0..7 ; elems 8..15 -> K = 16+half*8+0..7
      af[mt] = load_frag(&cA[rl + half * 8], &cA[rl + 16 + half * 8]);
    }
    #pragma unroll
    for (int nt = 0; nt < 4; ++nt) {
      const int rl = (wn + nt * 16 + (lane & 15)) * LPITCH;
      // B 32x16: elems 0..7 -> K = half*8+0..7 ; elems 8..15 -> K = 16+half*8+0..7
      bfv[nt] = load_frag(&cB[rl + half * 8], &cB[rl + 16 + half * 8]);
    }
    #pragma unroll
    for (int mt = 0; mt < 4; ++mt)
      #pragma unroll
      for (int nt = 0; nt < 4; ++nt)
        acc[mt * 4 + nt] = __builtin_amdgcn_wmma_f32_16x16x32_bf16(
            false, af[mt], false, bfv[nt], (short)0, acc[mt * 4 + nt], false, false);
  }

  // C layout: VGPR r -> row r + 8*half, col = lane&15 within each 16x16 tile
  #pragma unroll
  for (int mt = 0; mt < 4; ++mt)
    #pragma unroll
    for (int nt = 0; nt < 4; ++nt) {
      v8f a = acc[mt * 4 + nt];
      const int col = n0 + wn + nt * 16 + (lane & 15);
      #pragma unroll
      for (int r = 0; r < 8; ++r) {
        const int row = m0 + wm + mt * 16 + r + half * 8;
        if (OUT_BF16) Cb[(size_t)row * Nsz + col] = f2bf(a[r]);
        else          Cf[(size_t)row * Nsz + col] = a[r] + (HAS_BIAS ? bias[col] : 0.f);
      }
    }
}

// --------------------------- attention diagonal ----------------------------
// diag[b,n,i] = exp(s_ii)/sum_j exp(s_ij), s = Q K^T / sqrt(D).
// q[b,n,i,:] / k[b,n,j,:] are contiguous 128-elem chunks of the qkv matrix:
//   row = b*T + n*128 + (i>>4), col = (i&15)*128 (+2048 for k, +4096 for v).
// Block = 4 independent waves, each owns 64 rows; Q resident in registers
// (4 ksteps x 4 Mtiles of v16bf); j swept in 16-column tiles via WMMA.

__global__ __launch_bounds__(128)
void attn_diag_kernel(const unsigned short* __restrict__ qkvb, float* __restrict__ diag) {
  const int tid  = threadIdx.x;
  const int lane = tid & 31, half = (tid >> 4) & 1, wv = tid >> 5;
  const int bn = blockIdx.x;                 // b*16 + n
  const int b = bn >> 4, n = bn & 15;
  const int i0 = blockIdx.y * 256 + wv * 64;
  const float scale = 0.02209708691207961f;  // 1/sqrt(2048)

  v16bf qf[4][4];                            // [kstep][mtile]
  #pragma unroll
  for (int mt = 0; mt < 4; ++mt) {
    const int i = i0 + mt * 16 + (lane & 15);
    const unsigned short* qrow =
        qkvb + (size_t)(b * Tt + n * 128 + (i >> 4)) * D3 + (i & 15) * 128;
    #pragma unroll
    for (int ks = 0; ks < 4; ++ks)
      qf[ks][mt] = load_frag(qrow + ks * 32 + half * 8, qrow + ks * 32 + 16 + half * 8);
  }

  float psum[4][8], dcap[4][8];
  #pragma unroll
  for (int mt = 0; mt < 4; ++mt)
    #pragma unroll
    for (int r = 0; r < 8; ++r) { psum[mt][r] = 0.f; dcap[mt][r] = 0.f; }

  for (int jt = 0; jt < Tt / 16; ++jt) {
    const int j = jt * 16 + (lane & 15);
    const unsigned short* krow =
        qkvb + (size_t)(b * Tt + n * 128 + (j >> 4)) * D3 + 2048 + (j & 15) * 128;
    v16bf kf[4];
    #pragma unroll
    for (int ks = 0; ks < 4; ++ks)
      kf[ks] = load_frag(krow + ks * 32 + half * 8, krow + ks * 32 + 16 + half * 8);

    v8f acc[4] = {};
    #pragma unroll
    for (int ks = 0; ks < 4; ++ks)
      #pragma unroll
      for (int mt = 0; mt < 4; ++mt)
        acc[mt] = __builtin_amdgcn_wmma_f32_16x16x32_bf16(
            false, qf[ks][mt], false, kf[ks], (short)0, acc[mt], false, false);

    #pragma unroll
    for (int mt = 0; mt < 4; ++mt)
      #pragma unroll
      for (int r = 0; r < 8; ++r) {
        const float e = __expf(acc[mt][r] * scale);   // scores ~O(0.2): no max-sub needed
        psum[mt][r] += e;
        if (j == i0 + mt * 16 + r + half * 8) dcap[mt][r] = e;  // diagonal hit
      }
  }

  #pragma unroll
  for (int mt = 0; mt < 4; ++mt)
    #pragma unroll
    for (int r = 0; r < 8; ++r) {
      float s = psum[mt][r], d = dcap[mt][r];
      #pragma unroll
      for (int off = 8; off; off >>= 1) {
        s += __shfl_xor(s, off, 16);
        d += __shfl_xor(d, off, 16);
      }
      if ((lane & 15) == 0) {
        const int rowi = i0 + mt * 16 + r + half * 8;
        diag[(size_t)bn * Tt + rowi] = d / s;
      }
    }
}

// o[b,t,n*128+h] = diag[b,n,t] * v[b,n,t,h]  (stored bf16 for the proj GEMM)
__global__ __launch_bounds__(256)
void scale_v_kernel(const unsigned short* __restrict__ qkvb,
                    const float* __restrict__ diag,
                    unsigned short* __restrict__ ob) {
  const size_t idx = (size_t)blockIdx.x * 256 + threadIdx.x;   // < B*T*D exactly
  const int d = (int)(idx & 2047);
  const size_t bt = idx >> 11;
  const int t = (int)(bt & 2047);
  const int b = (int)(bt >> 11);
  const int n = d >> 7, h = d & 127;
  const float v  = bf2f(qkvb[(size_t)(b * Tt + n * 128 + (t >> 4)) * D3
                             + 4096 + (t & 15) * 128 + h]);
  const float dg = diag[(size_t)(b * 16 + n) * Tt + t];
  ob[idx] = f2bf(v * dg);
}

// ------------------------------- launcher ----------------------------------

extern "C" void kernel_launch(void* const* d_in, const int* in_sizes, int n_in,
                              void* d_out, int out_size, void* d_ws, size_t ws_size,
                              hipStream_t stream) {
  (void)in_sizes; (void)n_in; (void)out_size; (void)ws_size;
  const float* x      = (const float*)d_in[0];
  const float* w_qkv  = (const float*)d_in[1];
  const float* w_proj = (const float*)d_in[2];
  const float* b_proj = (const float*)d_in[3];
  float* out = (float*)d_out;

  char* ws = (char*)d_ws;
  size_t off = 0;
  auto alloc = [&](size_t bytes) -> void* {
    void* p = ws + off;
    off = (off + bytes + 255) & ~(size_t)255;
    return p;
  };
  unsigned short* xb     = (unsigned short*)alloc((size_t)8192 * 2048 * 2); // x bf16
  unsigned short* wqkvT  = (unsigned short*)alloc((size_t)6144 * 2048 * 2); // w_qkv^T bf16
  unsigned short* wprojT = (unsigned short*)alloc((size_t)2048 * 2048 * 2); // w_proj^T bf16
  unsigned short* qkvb   = (unsigned short*)alloc((size_t)8192 * 6144 * 2); // qkv bf16
  unsigned short* ob     = (unsigned short*)alloc((size_t)8192 * 2048 * 2); // o bf16
  float*          diag   = (float*)alloc((size_t)4 * 16 * 2048 * 4);        // diag fp32

  cvt_f32_bf16<<<65536, 256, 0, stream>>>(x, xb, 8192 * 2048);
  transpose_f32_to_bf16<<<dim3(192, 64), dim3(32, 8), 0, stream>>>(w_qkv, wqkvT, 2048, 6144);
  transpose_f32_to_bf16<<<dim3(64, 64),  dim3(32, 8), 0, stream>>>(w_proj, wprojT, 2048, 2048);

  // qkv = x @ w_qkv  (M=8192, K=2048, N=6144) -> bf16
  gemm_bf16_wmma<1, 0><<<dim3(48, 64), 128, 0, stream>>>(xb, wqkvT, nullptr, qkvb,
                                                         nullptr, 2048, 6144);
  // softmax-diagonal per (b, head)
  attn_diag_kernel<<<dim3(64, 8), 128, 0, stream>>>(qkvb, diag);
  // o = diag * v
  scale_v_kernel<<<65536, 256, 0, stream>>>(qkvb, diag, ob);
  // out = o @ w_proj + b_proj  (M=8192, K=2048, N=2048) -> fp32
  gemm_bf16_wmma<0, 1><<<dim3(16, 64), 128, 0, stream>>>(ob, wprojT, out, nullptr,
                                                         b_proj, 2048, 2048);
}
